// MultiHeadAttention_25477746000195
// MI455X (gfx1250) — matrix operand order repaired and checked by reading, not compiled, not run on hardware
//
#include <hip/hip_runtime.h>
#include <hip/hip_bf16.h>

typedef __attribute__((ext_vector_type(16))) _Float16 v16h;
typedef __attribute__((ext_vector_type(8)))  _Float16 v8h;
typedef __attribute__((ext_vector_type(8)))  float    v8f;

#define NB 2
#define NH 12
#define SL 2048
#define DM 768
#define DK 64
#define NEGV -1000000000.0f

// ---------------------------------------------------------------- WMMA helpers

__device__ __forceinline__ v8f wmma16(v16h a, v16h b, v8f c) {
  // D = A(16x32 f16) * B(32x16 f16) + C(16x16 f32)
  return __builtin_amdgcn_wmma_f32_16x16x32_f16(false, a, false, b, (short)0, c, false, false);
}

// A-fragment (16x32) from row-major half matrix.
// ISA layout: lanes0-15 row=lane, elems0-7 = K+0..7, elems8-15 = K+16..23;
//             lanes16-31 row=lane-16, elems0-7 = K+8..15, elems8-15 = K+24..31.
__device__ __forceinline__ v16h load_afrag_h(const _Float16* base, int ld, int row0, int k0) {
  const int lane = threadIdx.x & 31;
  const _Float16* p = base + (size_t)(row0 + (lane & 15)) * ld + k0 + ((lane < 16) ? 0 : 8);
  v8h lo = *(const v8h*)p;
  v8h hi = *(const v8h*)(p + 16);
  v16h a;
#pragma unroll
  for (int i = 0; i < 8; ++i) { a[i] = lo[i]; a[i + 8] = hi[i]; }
  return a;
}

// Same as above but converting from f32 source on the fly.
__device__ __forceinline__ v16h load_afrag_f32(const float* base, int ld, int row0, int k0) {
  const int lane = threadIdx.x & 31;
  const float* p = base + (size_t)(row0 + (lane & 15)) * ld + k0 + ((lane < 16) ? 0 : 8);
  v16h a;
#pragma unroll
  for (int i = 0; i < 8; ++i) { a[i] = (_Float16)p[i]; a[i + 8] = (_Float16)p[i + 16]; }
  return a;
}

// B-fragment (32x16) from column-major-per-N storage BT[n][k] (ld over k).
// ISA layout: lanes0-15 col=lane, elems0-7 = K+0..7, elems8-15 = K+16..23;
//             lanes16-31 col=lane-16, elems0-7 = K+8..15, elems8-15 = K+24..31.
__device__ __forceinline__ v16h load_bfrag_h(const _Float16* baseT, int ld, int n0, int k0) {
  const int lane = threadIdx.x & 31;
  const _Float16* p = baseT + (size_t)(n0 + (lane & 15)) * ld + k0 + ((lane < 16) ? 0 : 8);
  v8h lo = *(const v8h*)p;
  v8h hi = *(const v8h*)(p + 16);
  v16h b;
#pragma unroll
  for (int i = 0; i < 8; ++i) { b[i] = lo[i]; b[i + 8] = hi[i]; }
  return b;
}

// gfx1250 async copy: 16 bytes global -> LDS per lane (tracked by ASYNCcnt)
__device__ __forceinline__ void async_copy16(const _Float16* lds_dst, const _Float16* gsrc) {
  unsigned loff = (unsigned)(unsigned long long)(const void*)lds_dst;  // LDS flat addr low 32 = ds offset
  asm volatile("global_load_async_to_lds_b128 %0, %1, off"
               :: "v"(loff), "v"((unsigned long long)(const void*)gsrc)
               : "memory");
}

__device__ __forceinline__ void wait_async0() {
  asm volatile("s_wait_asynccnt 0x0" ::: "memory");
}

// ---------------------------------------------------------------- prep kernels

__global__ __launch_bounds__(256) void wtrans_kernel(const float* __restrict__ W,
                                                     _Float16* __restrict__ WT) {
  int idx = blockIdx.x * 256 + threadIdx.x;
  if (idx < DM * DM) {
    int n = idx / DM, k = idx - n * DM;
    WT[idx] = (_Float16)W[(size_t)k * DM + n];  // WT[n][k] = W[k][n]
  }
}

// vT[b][h][d][l] = (half)query[b][l][h*64+d]   (d-major V for PV B-fragments)
__global__ __launch_bounds__(256) void buildvt_kernel(const float* __restrict__ query,
                                                      _Float16* __restrict__ vT) {
  int idx = blockIdx.x * 256 + threadIdx.x;
  if (idx < NB * NH * DK * SL) {
    int l = idx & (SL - 1);
    int rest = idx >> 11;
    int d = rest & 63;
    int rest2 = rest >> 6;
    int h = rest2 % NH, b = rest2 / NH;
    vT[idx] = (_Float16)query[((size_t)b * SL + l) * DM + h * DK + d];
  }
}

// ---------------------------------------------------------------- projections
// out_h[b][h][l][d] = (half)( X[(b,l)] @ W + bias )   via WMMA f16 GEMM
__global__ __launch_bounds__(256) void proj_kernel(const float* __restrict__ X,
                                                   const _Float16* __restrict__ WT,
                                                   const float* __restrict__ bias,
                                                   _Float16* __restrict__ outh) {
  const int w = threadIdx.x >> 5, lane = threadIdx.x & 31;
  const int row0 = (blockIdx.y * 8 + w) * 16;   // 0..4080 over B*L rows
  const int n0 = blockIdx.x * 16;               // 0..752 over D cols
  v8f c = {};
  for (int k0 = 0; k0 < DM; k0 += 32) {
    v16h a = load_afrag_f32(X, DM, row0, k0);
    v16h bf = load_bfrag_h(WT, DM, n0, k0);
    c = wmma16(a, bf, c);
  }
  const int n = n0 + (lane & 15);
  const float bn = bias[n];
  const int hh = n >> 6, d = n & 63;
#pragma unroll
  for (int r = 0; r < 8; ++r) {
    int gr = row0 + r + ((lane < 16) ? 0 : 8);
    int b = gr >> 11, l = gr & (SL - 1);
    outh[(((size_t)(b * NH + hh)) * SL + l) * DK + d] = (_Float16)(c[r] + bn);
  }
}

// ---------------------------------------------------------------- Sinkhorn (pi)

__device__ __forceinline__ float block_reduce(float x, bool domax,
                                              float* red_s, float* bcast) {
  const int lane = threadIdx.x & 31, w = threadIdx.x >> 5;
#pragma unroll
  for (int d = 16; d > 0; d >>= 1) {
    float o = __shfl_xor(x, d, 32);
    x = domax ? fmaxf(x, o) : (x + o);
  }
  if (lane == 0) red_s[w] = x;
  __syncthreads();
  if (threadIdx.x == 0) {
    float t = red_s[0];
    for (int i = 1; i < 8; ++i) t = domax ? fmaxf(t, red_s[i]) : (t + red_s[i]);
    *bcast = t;
  }
  __syncthreads();
  float r = *bcast;
  __syncthreads();
  return r;
}

__global__ __launch_bounds__(256) void sinkhorn_kernel(const float* __restrict__ query,
                                                       const float* __restrict__ aspect,
                                                       const int* __restrict__ mask,
                                                       const float* __restrict__ w_mu,
                                                       const float* __restrict__ b_mu,
                                                       float* __restrict__ pi) {
  const int h = blockIdx.x, b = blockIdx.y;
  const int tid = threadIdx.x;
  __shared__ float asp_s[64], wmu_s[64], red_s[8];
  __shared__ float bcast_s;
  if (tid < 64) { asp_s[tid] = aspect[b * DM + h * DK + tid]; wmu_s[tid] = w_mu[tid]; }
  __syncthreads();
  float na2 = 0.0f;
  for (int d = 0; d < 64; ++d) na2 += asp_s[d] * asp_s[d];
  const float na = fmaxf(sqrtf(na2), 1e-12f);
  const float bmu = b_mu[0];

  float Kl[8], mu_l[8], u_l[8];
  float mymax = -3.0e38f;
#pragma unroll
  for (int i = 0; i < 8; ++i) {
    int l = i * 256 + tid;
    const float* sp = query + ((size_t)b * SL + l) * DM + h * DK;
    float sq = 0, dw = 0, da = 0;
    for (int d = 0; d < 64; ++d) {
      float x = sp[d];
      sq += x * x; dw += x * wmu_s[d]; da += x * asp_s[d];
    }
    float cosv = da / (fmaxf(sqrtf(sq), 1e-12f) * na);
    Kl[i] = __expf(cosv - 1.0f);                 // exp(-cost/OT_EPS), OT_EPS=1
    float lg = dw + bmu;
    if (mask[b * SL + l] == 0) lg = NEGV;
    mu_l[i] = lg;
    mymax = fmaxf(mymax, lg);
  }
  float gmax = block_reduce(mymax, true, red_s, &bcast_s);
  float ps = 0.0f;
#pragma unroll
  for (int i = 0; i < 8; ++i) { mu_l[i] = __expf(mu_l[i] - gmax); ps += mu_l[i]; }
  float gsum = block_reduce(ps, false, red_s, &bcast_s);
  float invs = 1.0f / gsum;
#pragma unroll
  for (int i = 0; i < 8; ++i) mu_l[i] *= invs;

  float v = 1.0f;                                 // nu == 1 (softmax over axis of len 1)
  for (int it = 0; it < 50; ++it) {
    float sp2 = 0.0f;
#pragma unroll
    for (int i = 0; i < 8; ++i) {
      u_l[i] = mu_l[i] / (Kl[i] * v + 1e-8f);
      sp2 += Kl[i] * u_l[i];
    }
    float S = block_reduce(sp2, false, red_s, &bcast_s);
    v = 1.0f / (S + 1e-8f);
  }
#pragma unroll
  for (int i = 0; i < 8; ++i)
    pi[(size_t)(b * NH + h) * SL + i * 256 + tid] = u_l[i] * Kl[i] * v;
}

// ---------------------------------------------------------------- fused flash attention
// Per block: 128 query rows (16/warp). K/V tiles (32 cols) staged to LDS via
// gfx1250 async copies, double buffered; scores + online softmax + P*V via WMMA.

#define KPAD 72   // 32x64 K tile rows padded 64->72 halves (bank-conflict-free)
#define VPAD 40   // 64x32 V tile rows padded 32->40 halves
#define PPAD 40   // P relayout rows padded 32->40 halves

__global__ __launch_bounds__(256) void attn_fused_kernel(
    const _Float16* __restrict__ qh, const _Float16* __restrict__ kh,
    const _Float16* __restrict__ vT, const float* __restrict__ shortb,
    const int* __restrict__ mask, const float* __restrict__ query,
    const float* __restrict__ pi, const float* __restrict__ bias_m,
    float* __restrict__ out) {
  const int tid = threadIdx.x;
  const int w = tid >> 5;
  const int lane = tid & 31;
  const int h = blockIdx.y, b = blockIdx.z;
  const int l0 = blockIdx.x * 128 + w * 16;       // each warp owns 16 query rows

  const _Float16* qbh = qh + (size_t)(b * NH + h) * SL * DK;
  const _Float16* kbh = kh + (size_t)(b * NH + h) * SL * DK;
  const _Float16* vbh = vT + (size_t)(b * NH + h) * DK * SL;
  const float* sbh = shortb + (size_t)(b * NH + h) * SL * SL;
  const int* mb = mask + b * SL;
  const float bm = bias_m[0];

  __shared__ __attribute__((aligned(16))) _Float16 plds[8][16][PPAD];
  __shared__ __attribute__((aligned(16))) _Float16 kbuf[2][32][KPAD];
  __shared__ __attribute__((aligned(16))) _Float16 vbuf[2][64][VPAD];

  // cooperative async staging: each of 256 threads copies 16B of each tile
  const int krow = tid >> 3, kch = (tid & 7) * 8;       // K: 32 rows x 64 halves
  const int vrow = tid >> 2, vch = (tid & 3) * 8;       // V: 64 rows x 32 halves
  auto issue_tiles = [&](int bufn, int c0) {
    async_copy16(&kbuf[bufn][krow][kch], kbh + (size_t)(c0 + krow) * DK + kch);
    async_copy16(&vbuf[bufn][vrow][vch], vbh + (size_t)vrow * SL + c0 + vch);
  };

  const v16h aq0 = load_afrag_h(qbh, DK, l0, 0);
  const v16h aq1 = load_afrag_h(qbh, DK, l0, 32);

  float mrow[8], lrow[8];
  v8f o0 = {}, o1 = {}, o2 = {}, o3 = {};
#pragma unroll
  for (int r = 0; r < 8; ++r) { mrow[r] = -1e30f; lrow[r] = 0.0f; }

  const int halfoff = (lane < 16) ? 0 : 8;
  const int colb = lane & 15;

  issue_tiles(0, 0);

  for (int c0 = 0, it = 0; c0 < SL; c0 += 32, ++it) {
    const int cur = it & 1;
    wait_async0();                                // this wave's share landed in LDS
    __syncthreads();                              // everyone's share visible
    if (c0 + 32 < SL) issue_tiles(cur ^ 1, c0 + 32);

    const _Float16* kt = &kbuf[cur][0][0];
    const _Float16* vt = &vbuf[cur][0][0];

    // ---- scores S = (Q K^T) for two 16-col tiles, K-dim = dk = 64
    v8f s0 = {}, s1 = {};
    s0 = wmma16(aq0, load_bfrag_h(kt, KPAD, 0, 0), s0);
    s0 = wmma16(aq1, load_bfrag_h(kt, KPAD, 0, 32), s0);
    s1 = wmma16(aq0, load_bfrag_h(kt, KPAD, 16, 0), s1);
    s1 = wmma16(aq1, load_bfrag_h(kt, KPAD, 16, 32), s1);

    const int n0c = c0 + colb, n1c = c0 + 16 + colb;
    const bool mz0 = (mb[n0c] == 0), mz1 = (mb[n1c] == 0);

    float mnew[8], scl[8];
#pragma unroll
    for (int r = 0; r < 8; ++r) {
      const float* srow = sbh + (size_t)(l0 + r + halfoff) * SL;
      float a0 = s0[r] * 0.125f; if (mz0) a0 = NEGV;
      float a1 = s1[r] * 0.125f; if (mz1) a1 = NEGV;
      a0 += srow[n0c];
      a1 += srow[n1c];
      s0[r] = a0; s1[r] = a1;
      float t = fmaxf(a0, a1);                    // row max across 16 lanes
#pragma unroll
      for (int d = 1; d < 16; d <<= 1) t = fmaxf(t, __shfl_xor(t, d, 32));
      mnew[r] = fmaxf(mrow[r], t);
      scl[r] = __expf(mrow[r] - mnew[r]);
      mrow[r] = mnew[r];
    }
#pragma unroll
    for (int r = 0; r < 8; ++r) {
      float p0 = __expf(s0[r] - mnew[r]);
      float p1 = __expf(s1[r] - mnew[r]);
      float rs = p0 + p1;
#pragma unroll
      for (int d = 1; d < 16; d <<= 1) rs += __shfl_xor(rs, d, 32);
      lrow[r] = lrow[r] * scl[r] + rs;
      o0[r] *= scl[r]; o1[r] *= scl[r]; o2[r] *= scl[r]; o3[r] *= scl[r];
      const int M = r + halfoff;                  // relayout P: C-frag -> LDS tile
      plds[w][M][colb] = (_Float16)p0;
      plds[w][M][16 + colb] = (_Float16)p1;
    }
    __builtin_amdgcn_wave_barrier();              // same-wave LDS is in-order; pin compiler
    const v16h ap = load_afrag_h(&plds[w][0][0], PPAD, 0, 0);

    // ---- O += P * V  (K'=32 rows of V, 64 output cols)
    o0 = wmma16(ap, load_bfrag_h(vt, VPAD, 0,  0), o0);
    o1 = wmma16(ap, load_bfrag_h(vt, VPAD, 16, 0), o1);
    o2 = wmma16(ap, load_bfrag_h(vt, VPAD, 32, 0), o2);
    o3 = wmma16(ap, load_bfrag_h(vt, VPAD, 48, 0), o3);

    if (c0 + 32 < SL) {                           // stream-prefetch next short tile
      const float* pf = sbh + (size_t)(l0 + colb) * SL + c0 + 32 + ((lane < 16) ? 0 : 16);
      __builtin_prefetch(pf, 0, 0);
    }
  }

  // ---- epilogue: normalize, blend with Sinkhorn transport plan
#pragma unroll
  for (int r = 0; r < 8; ++r) {
    const int l = l0 + r + halfoff;
    const float inv = 1.0f / lrow[r];
    const float pv = 0.2f * pi[(size_t)(b * NH + h) * SL + l];
    const float* qr = query + ((size_t)b * SL + l) * DM + h * DK;
    float* orow = out + ((size_t)(b * NH + h) * SL + l) * DK;
    const float acc[4] = { o0[r], o1[r], o2[r], o3[r] };
#pragma unroll
    for (int t = 0; t < 4; ++t) {
      int d = t * 16 + colb;
      orow[d] = 0.8f * (acc[t] * inv) + pv * qr[d] + bm;
    }
  }
}

// ---------------------------------------------------------------- launch

extern "C" void kernel_launch(void* const* d_in, const int* in_sizes, int n_in,
                              void* d_out, int out_size, void* d_ws, size_t ws_size,
                              hipStream_t stream) {
  const float* query  = (const float*)d_in[0];
  const float* key    = (const float*)d_in[1];
  const float* short_ = (const float*)d_in[2];
  const float* aspect = (const float*)d_in[3];
  const int*   mask   = (const int*)d_in[4];
  const float* Wq     = (const float*)d_in[5];
  const float* bq     = (const float*)d_in[6];
  const float* Wk     = (const float*)d_in[7];
  const float* bk     = (const float*)d_in[8];
  const float* w_mu   = (const float*)d_in[9];
  const float* b_mu   = (const float*)d_in[10];
  const float* bias_m = (const float*)d_in[11];
  float* out = (float*)d_out;

  char* ws = (char*)d_ws;
  const size_t HEADS_HALF = (size_t)NB * NH * SL * DK * sizeof(_Float16);  // 6 MB
  const size_t W_HALF = (size_t)DM * DM * sizeof(_Float16);                // 1.125 MB
  _Float16* qh  = (_Float16*)(ws);
  _Float16* kh  = (_Float16*)(ws + HEADS_HALF);
  _Float16* vT  = (_Float16*)(ws + 2 * HEADS_HALF);
  _Float16* wqT = (_Float16*)(ws + 3 * HEADS_HALF);
  _Float16* wkT = (_Float16*)(ws + 3 * HEADS_HALF + W_HALF);
  float*    pi  = (float*)(ws + 3 * HEADS_HALF + 2 * W_HALF);

  wtrans_kernel<<<(DM * DM + 255) / 256, 256, 0, stream>>>(Wq, wqT);
  wtrans_kernel<<<(DM * DM + 255) / 256, 256, 0, stream>>>(Wk, wkT);
  buildvt_kernel<<<(NB * NH * DK * SL + 255) / 256, 256, 0, stream>>>(query, vT);

  proj_kernel<<<dim3(DM / 16, (NB * SL) / 128), 256, 0, stream>>>(query, wqT, bq, qh);
  proj_kernel<<<dim3(DM / 16, (NB * SL) / 128), 256, 0, stream>>>(key,   wkT, bk, kh);

  sinkhorn_kernel<<<dim3(NH, NB), 256, 0, stream>>>(query, aspect, mask, w_mu, b_mu, pi);

  attn_fused_kernel<<<dim3(SL / 128, NH, NB), 256, 0, stream>>>(
      qh, kh, vT, short_, mask, query, pi, bias_m, out);
}
